// SelectiveChannelMixer_72232759984890
// MI455X (gfx1250) — matrix operand order repaired and checked by reading, not compiled, not run on hardware
//
#include <hip/hip_runtime.h>
#include <hip/hip_bf16.h>

typedef __attribute__((ext_vector_type(16))) __bf16 v16bf;
typedef __attribute__((ext_vector_type(8)))  __bf16 v8bf;
typedef __attribute__((ext_vector_type(8)))  float  v8f;

// ---------------- problem constants ----------------
constexpr int kBN    = 128;
constexpr int kSeq   = 64;
constexpr int kDM    = 256;   // d_model
constexpr int kDI    = 512;   // d_inner
constexpr int kRows  = kBN * kSeq;      // 8192
constexpr int kK2    = 2 * kDI;         // 1024
constexpr int kDbc   = 48;              // dt_rank + 2*d_state
constexpr int kState = 16;

// ---------------- workspace layout (bytes, all 256B aligned) ----------------
constexpr size_t OFF_XN     = 0;
constexpr size_t OFF_WIN_F  = OFF_XN     + (size_t)kRows * kDM * 2;
constexpr size_t OFF_WIN_B  = OFF_WIN_F  + (size_t)kK2   * kDM * 2;
constexpr size_t OFF_WXP_F  = OFF_WIN_B  + (size_t)kK2   * kDM * 2;
constexpr size_t OFF_WXP_B  = OFF_WXP_F  + (size_t)kDbc  * kDI * 2;
constexpr size_t OFF_WOUT_F = OFF_WXP_B  + (size_t)kDbc  * kDI * 2;
constexpr size_t OFF_WOUT_B = OFF_WOUT_F + (size_t)kDM   * kDI * 2;
constexpr size_t OFF_XZ_F   = OFF_WOUT_B + (size_t)kDM   * kDI * 2;
constexpr size_t OFF_XZ_B   = OFF_XZ_F   + (size_t)kRows * kK2 * 2;
constexpr size_t OFF_XS_F   = OFF_XZ_B   + (size_t)kRows * kK2 * 2;
constexpr size_t OFF_XS_B   = OFF_XS_F   + (size_t)kRows * kDI * 2;
constexpr size_t OFF_DBC_F  = OFF_XS_B   + (size_t)kRows * kDI * 2;
constexpr size_t OFF_DBC_B  = OFF_DBC_F  + (size_t)kRows * kDbc * 4;
constexpr size_t OFF_Y_F    = OFF_DBC_B  + (size_t)kRows * kDbc * 4;
constexpr size_t OFF_Y_B    = OFF_Y_F    + (size_t)kRows * kDI * 2;

// ---------------- LayerNorm -> bf16 (one wave32 per row) ----------------
__global__ __launch_bounds__(256) void ln_kernel(const float* __restrict__ x,
                                                 const float* __restrict__ g,
                                                 const float* __restrict__ be,
                                                 __bf16* __restrict__ xn) {
  int wave = threadIdx.x >> 5, lane = threadIdx.x & 31;
  int row  = blockIdx.x * 8 + wave;
  const float* xr = x + (size_t)row * kDM;
  float v[8], s = 0.f, s2 = 0.f;
#pragma unroll
  for (int i = 0; i < 8; i++) {
    v[i] = xr[lane + i * 32];
    s += v[i]; s2 += v[i] * v[i];
  }
#pragma unroll
  for (int o = 16; o >= 1; o >>= 1) {
    s  += __shfl_xor(s,  o, 32);
    s2 += __shfl_xor(s2, o, 32);
  }
  float mu  = s / kDM;
  float var = s2 / kDM - mu * mu;
  float inv = rsqrtf(var + 1e-5f);
#pragma unroll
  for (int i = 0; i < 8; i++) {
    int c = lane + i * 32;
    xn[(size_t)row * kDM + c] = (__bf16)(((v[i] - mu) * inv) * g[c] + be[c]);
  }
}

// ---------------- fp32 -> bf16 weight convert ----------------
__global__ void f2bf_kernel(const float* __restrict__ in, __bf16* __restrict__ out, int n) {
  int i = blockIdx.x * blockDim.x + threadIdx.x;
  if (i < n) out[i] = (__bf16)in[i];
}

// ---- A-fragment loader: 16x32 bf16 tile, row-major source, per-lane gather ----
// lanes 0-15 hold K {0..15}; lanes 16-31 hold K {16..31} of row lane&15
__device__ __forceinline__ v16bf load_a_frag(const __bf16* __restrict__ arow, int k, int hi) {
  v8bf a_lo = *(const v8bf*)(arow + k + hi * 16);
  v8bf a_hi = *(const v8bf*)(arow + k + hi * 16 + 8);
  v16bf a;
#pragma unroll
  for (int i = 0; i < 8; i++) { a[i] = a_lo[i]; a[8 + i] = a_hi[i]; }
  return a;
}

// ---------------- WMMA GEMM: C[M,N] = A[M,K] * W[N,K]^T ----------------
// Each wave computes a 16 x (16*NT) strip: A fragment loaded once per k-step,
// streamed against NT B fragments (4x less A traffic at NT=4).
// 8 waves / block cover 128 rows at one N strip.
template <int NT, bool STORE_BF16>
__global__ __launch_bounds__(256) void gemm_wmma(const __bf16* __restrict__ A,
                                                 const __bf16* __restrict__ W,
                                                 void* __restrict__ Cout,
                                                 int N, int K) {
  int wave = threadIdx.x >> 5, lane = threadIdx.x & 31;
  int m0 = (blockIdx.y * 8 + wave) * 16;
  int n0 = blockIdx.x * (16 * NT);
  int lr = lane & 15;
  int hi = lane >> 4;  // 0: lanes 0-15, 1: lanes 16-31

  const __bf16* arow = A + (size_t)(m0 + lr) * K;
  const __bf16* wrow[NT];
#pragma unroll
  for (int nt = 0; nt < NT; nt++)
    wrow[nt] = W + (size_t)(n0 + nt * 16 + lr) * K;

  v8f acc[NT];
#pragma unroll
  for (int nt = 0; nt < NT; nt++) acc[nt] = {};

  for (int k = 0; k < K; k += 32) {
    v16bf a = load_a_frag(arow, k, hi);
#pragma unroll
    for (int nt = 0; nt < NT; nt++) {
      // B 32x16 bf16: column n = lane&15; lanes0-15 K k..k+15, lanes16-31 K k+16..k+31
      v16bf b = *(const v16bf*)(wrow[nt] + k + hi * 16);
      acc[nt] = __builtin_amdgcn_wmma_f32_16x16x32_bf16(false, a, false, b, (short)0,
                                                        acc[nt], false, false);
    }
  }
  int rbase = m0 + hi * 8;
#pragma unroll
  for (int nt = 0; nt < NT; nt++) {
    int col = n0 + nt * 16 + lr;
    if (STORE_BF16) {
      __bf16* C = (__bf16*)Cout;
#pragma unroll
      for (int v = 0; v < 8; v++) C[(size_t)(rbase + v) * N + col] = (__bf16)acc[nt][v];
    } else {
      float* C = (float*)Cout;
#pragma unroll
      for (int v = 0; v < 8; v++) C[(size_t)(rbase + v) * N + col] = acc[nt][v];
    }
  }
}

// ---------------- fused out_proj: Out = Yf*Wf^T + Yb*Wb^T + X ----------------
// Same 16x64 strip-per-wave structure, two K passes into the same accumulators.
__global__ __launch_bounds__(256) void out_fused_wmma(const __bf16* __restrict__ Yf,
                                                      const __bf16* __restrict__ Wf,
                                                      const __bf16* __restrict__ Yb,
                                                      const __bf16* __restrict__ Wb,
                                                      const float* __restrict__ X,
                                                      float* __restrict__ Out) {
  constexpr int NT = 4;
  int wave = threadIdx.x >> 5, lane = threadIdx.x & 31;
  int m0 = (blockIdx.y * 8 + wave) * 16;
  int n0 = blockIdx.x * (16 * NT);
  int lr = lane & 15;
  int hi = lane >> 4;

  v8f acc[NT];
#pragma unroll
  for (int nt = 0; nt < NT; nt++) acc[nt] = {};

#pragma unroll
  for (int pass = 0; pass < 2; pass++) {
    const __bf16* arow = (pass == 0 ? Yf : Yb) + (size_t)(m0 + lr) * kDI;
    const __bf16* wbase = (pass == 0 ? Wf : Wb);
    const __bf16* wrow[NT];
#pragma unroll
    for (int nt = 0; nt < NT; nt++)
      wrow[nt] = wbase + (size_t)(n0 + nt * 16 + lr) * kDI;
    for (int k = 0; k < kDI; k += 32) {
      v16bf a = load_a_frag(arow, k, hi);
#pragma unroll
      for (int nt = 0; nt < NT; nt++) {
        v16bf b = *(const v16bf*)(wrow[nt] + k + hi * 16);
        acc[nt] = __builtin_amdgcn_wmma_f32_16x16x32_bf16(false, a, false, b, (short)0,
                                                          acc[nt], false, false);
      }
    }
  }
  int rbase = m0 + hi * 8;
#pragma unroll
  for (int nt = 0; nt < NT; nt++) {
    int col = n0 + nt * 16 + lr;
#pragma unroll
    for (int v = 0; v < 8; v++) {
      size_t idx = (size_t)(rbase + v) * kDM + col;
      Out[idx] = acc[nt][v] + X[idx];
    }
  }
}

// ---------------- depthwise causal conv(4) + bias + SiLU ----------------
// DIR==0: taps at t-3..t (forward); DIR==1: taps at t..t+3 (reversed sequence)
template <int DIR>
__global__ __launch_bounds__(256) void conv_silu_kernel(const __bf16* __restrict__ xz,
                                                        const float* __restrict__ cw,
                                                        const float* __restrict__ cb,
                                                        __bf16* __restrict__ xs_out) {
  int idx = blockIdx.x * blockDim.x + threadIdx.x;  // row*kDI + ch
  int ch  = idx & (kDI - 1);
  int row = idx >> 9;
  int t    = row & (kSeq - 1);
  int base = row - t;  // b*kSeq
  float acc = cb[ch];
#pragma unroll
  for (int j = 0; j < 4; j++) {
    int tj = (DIR == 0) ? (t - 3 + j) : (t + 3 - j);
    if (tj >= 0 && tj < kSeq)
      acc += cw[ch * 4 + j] * (float)xz[(size_t)(base + tj) * kK2 + ch];
  }
  float s = acc / (1.f + expf(-acc));  // silu
  xs_out[idx] = (__bf16)s;
}

// ---------------- fused dt_proj + softplus + selective scan + gating ----------------
// One thread per (batch, channel); block = 256 channels of one batch element.
template <int DIR>
__global__ __launch_bounds__(256) void scan_kernel(const __bf16* __restrict__ xz,
                                                   const __bf16* __restrict__ xs,
                                                   const float* __restrict__ dbc,
                                                   const float* __restrict__ dtw,
                                                   const float* __restrict__ dtb,
                                                   const float* __restrict__ A_log,
                                                   const float* __restrict__ Dv,
                                                   __bf16* __restrict__ y_out) {
  __shared__ float sm[kDbc];
  int b  = blockIdx.x >> 1;
  int ch = ((blockIdx.x & 1) << 8) + threadIdx.x;

  float Arow[kState], Wrow[kState], h[kState];
#pragma unroll
  for (int s = 0; s < kState; s++) {
    Arow[s] = -expf(A_log[ch * kState + s]);
    Wrow[s] = dtw[ch * kState + s];
    h[s] = 0.f;
  }
  float bias = dtb[ch], Dch = Dv[ch];

  for (int step = 0; step < kSeq; step++) {
    int t   = (DIR == 0) ? step : (kSeq - 1 - step);
    int row = b * kSeq + t;
    __syncthreads();
    if (threadIdx.x < kDbc) sm[threadIdx.x] = dbc[(size_t)row * kDbc + threadIdx.x];
    __syncthreads();

    float xv  = (float)xs[(size_t)row * kDI + ch];
    float dtp = bias;
#pragma unroll
    for (int s = 0; s < kState; s++) dtp += sm[s] * Wrow[s];
    float dt = (dtp > 20.f) ? dtp : log1pf(expf(dtp));  // softplus

    float y = 0.f;
#pragma unroll
    for (int s = 0; s < kState; s++) {
      float dA = expf(dt * Arow[s]);
      h[s] = dA * h[s] + dt * sm[kState + s] * xv;   // B part
      y += h[s] * sm[2 * kState + s];                // C part
    }
    y += xv * Dch;
    float z = (float)xz[(size_t)row * kK2 + kDI + ch];
    y *= z / (1.f + expf(-z));  // * silu(z)
    y_out[(size_t)row * kDI + ch] = (__bf16)y;
  }
}

// ---------------- launch ----------------
extern "C" void kernel_launch(void* const* d_in, const int* in_sizes, int n_in,
                              void* d_out, int out_size, void* d_ws, size_t ws_size,
                              hipStream_t stream) {
  const float* x     = (const float*)d_in[0];
  const float* ln_g  = (const float*)d_in[1];
  const float* ln_b  = (const float*)d_in[2];
  const float* f_in  = (const float*)d_in[3];
  const float* b_in  = (const float*)d_in[4];
  const float* f_cw  = (const float*)d_in[5];
  const float* b_cw  = (const float*)d_in[6];
  const float* f_cb  = (const float*)d_in[7];
  const float* b_cb  = (const float*)d_in[8];
  const float* f_xp  = (const float*)d_in[9];
  const float* b_xp  = (const float*)d_in[10];
  const float* f_dtw = (const float*)d_in[11];
  const float* b_dtw = (const float*)d_in[12];
  const float* f_dtb = (const float*)d_in[13];
  const float* b_dtb = (const float*)d_in[14];
  const float* f_al  = (const float*)d_in[15];
  const float* b_al  = (const float*)d_in[16];
  const float* f_Dv  = (const float*)d_in[17];
  const float* b_Dv  = (const float*)d_in[18];
  const float* f_ow  = (const float*)d_in[19];
  const float* b_ow  = (const float*)d_in[20];

  char* ws = (char*)d_ws;
  __bf16* xn     = (__bf16*)(ws + OFF_XN);
  __bf16* win_f  = (__bf16*)(ws + OFF_WIN_F);
  __bf16* win_b  = (__bf16*)(ws + OFF_WIN_B);
  __bf16* wxp_f  = (__bf16*)(ws + OFF_WXP_F);
  __bf16* wxp_b  = (__bf16*)(ws + OFF_WXP_B);
  __bf16* wout_f = (__bf16*)(ws + OFF_WOUT_F);
  __bf16* wout_b = (__bf16*)(ws + OFF_WOUT_B);
  __bf16* xz_f   = (__bf16*)(ws + OFF_XZ_F);
  __bf16* xz_b   = (__bf16*)(ws + OFF_XZ_B);
  __bf16* xs_f   = (__bf16*)(ws + OFF_XS_F);
  __bf16* xs_b   = (__bf16*)(ws + OFF_XS_B);
  float*  dbc_f  = (float*)(ws + OFF_DBC_F);
  float*  dbc_b  = (float*)(ws + OFF_DBC_B);
  __bf16* y_f    = (__bf16*)(ws + OFF_Y_F);
  __bf16* y_b    = (__bf16*)(ws + OFF_Y_B);

  // 1. LayerNorm -> bf16
  ln_kernel<<<kRows / 8, 256, 0, stream>>>(x, ln_g, ln_b, xn);

  // 2. weight converts
  f2bf_kernel<<<(kK2 * kDM + 255) / 256, 256, 0, stream>>>(f_in, win_f, kK2 * kDM);
  f2bf_kernel<<<(kK2 * kDM + 255) / 256, 256, 0, stream>>>(b_in, win_b, kK2 * kDM);
  f2bf_kernel<<<(kDbc * kDI + 255) / 256, 256, 0, stream>>>(f_xp, wxp_f, kDbc * kDI);
  f2bf_kernel<<<(kDbc * kDI + 255) / 256, 256, 0, stream>>>(b_xp, wxp_b, kDbc * kDI);
  f2bf_kernel<<<(kDM * kDI + 255) / 256, 256, 0, stream>>>(f_ow, wout_f, kDM * kDI);
  f2bf_kernel<<<(kDM * kDI + 255) / 256, 256, 0, stream>>>(b_ow, wout_b, kDM * kDI);

  // 3. in_proj GEMMs (8192x256 * 256x1024), 16x64 strip per wave
  gemm_wmma<4, true><<<dim3(kK2 / 64, kRows / 128), 256, 0, stream>>>(xn, win_f, xz_f,
                                                                      kK2, kDM);
  gemm_wmma<4, true><<<dim3(kK2 / 64, kRows / 128), 256, 0, stream>>>(xn, win_b, xz_b,
                                                                      kK2, kDM);

  // 4. depthwise conv + SiLU
  conv_silu_kernel<0><<<(kRows * kDI) / 256, 256, 0, stream>>>(xz_f, f_cw, f_cb, xs_f);
  conv_silu_kernel<1><<<(kRows * kDI) / 256, 256, 0, stream>>>(xz_b, b_cw, b_cb, xs_b);

  // 5. x_proj GEMMs (8192x512 * 512x48), whole N=48 strip per wave
  gemm_wmma<3, false><<<dim3(1, kRows / 128), 256, 0, stream>>>(xs_f, wxp_f, dbc_f,
                                                                kDbc, kDI);
  gemm_wmma<3, false><<<dim3(1, kRows / 128), 256, 0, stream>>>(xs_b, wxp_b, dbc_b,
                                                                kDbc, kDI);

  // 6. fused dt_proj + softplus + scan + gate
  scan_kernel<0><<<kBN * 2, 256, 0, stream>>>(xz_f, xs_f, dbc_f, f_dtw, f_dtb, f_al, f_Dv, y_f);
  scan_kernel<1><<<kBN * 2, 256, 0, stream>>>(xz_b, xs_b, dbc_b, b_dtw, b_dtb, b_al, b_Dv, y_b);

  // 7. fused out_proj + sum + residual (16x64 strip per wave)
  out_fused_wmma<<<dim3(kDM / 64, kRows / 128), 256, 0, stream>>>(y_f, wout_f, y_b, wout_b,
                                                                  x, (float*)d_out);
}
